// Decoder_1047972020583
// MI455X (gfx1250) — matrix operand order repaired and checked by reading, not compiled, not run on hardware
//
#include <hip/hip_runtime.h>
#include <hip/hip_bf16.h>
#include <math.h>

typedef __bf16 bf16_t;
typedef __attribute__((ext_vector_type(16))) __bf16 v16bf;
typedef __attribute__((ext_vector_type(8)))  float  v8f;

#define LANE (threadIdx.x & 31)
#define WAVE (threadIdx.x >> 5)

// ---------------- WMMA helpers (CDNA5 wave32, 16x16x32 bf16, f32 accum) ----

__device__ __forceinline__ v8f wmma_bf16(v16bf a, v16bf b, v8f c) {
  // (neg_a, A, neg_b, B, c_mod, C, reuse_a, reuse_b)
  return __builtin_amdgcn_wmma_f32_16x16x32_bf16(false, a, false, b, (short)0, c,
                                                 false, false);
}

// gfx1250 hardware transcendental tanh (V_TANH_F32). TRANS-op hazard rule:
// 1 independent op or V_NOP after a trans op before its result is consumed.
__device__ __forceinline__ float fast_tanh(float x) {
  float r;
  asm volatile("v_tanh_f32 %0, %1\n\tv_nop" : "=v"(r) : "v"(x));
  return r;
}

// A-matrix 16x32 bf16 fragment from row-major [.. x lda] at tile (m0,k0).
// Lanes 0-15: row m0+lane, K = {k0..k0+7, k0+16..k0+23}
// Lanes 16-31: row m0+(lane-16), K = {k0+8..k0+15, k0+24..k0+31}
__device__ __forceinline__ v16bf load_a(const bf16_t* A, int lda, int m0, int k0) {
  int row = m0 + (LANE & 15);
  int kk  = k0 + ((LANE >> 4) << 3);
  const bf16_t* p = A + (size_t)row * lda + kk;
  v16bf r;
#pragma unroll
  for (int i = 0; i < 8; ++i) { r[i] = p[i]; r[8 + i] = p[16 + i]; }
  return r;
}

// B-matrix 32x16 bf16 fragment. Weight stored row-major (N_out, K), so B col n
// == W row n. Lanes 0-15: col n0+lane, K={k0..k0+7, k0+16..k0+23} ; lanes 16-31: K={k0+8..k0+15, k0+24..k0+31}.
__device__ __forceinline__ v16bf load_b(const bf16_t* W, int ldw, int n0, int k0) {
  int n  = n0 + (LANE & 15);
  int kk = k0 + ((LANE >> 4) << 3);
  const bf16_t* p = W + (size_t)n * ldw + kk;
  v16bf r;
#pragma unroll
  for (int i = 0; i < 8; ++i) { r[i] = p[i]; r[8 + i] = p[16 + i]; }
  return r;
}

// C/D 16x16 f32: element i -> (m0 + ((lane>>4)<<3) + i, n0 + (lane&15))
__device__ __forceinline__ void store_tile(float* C, int ldc, int m0, int n0,
                                           v8f acc, const float* bias) {
  int n  = n0 + (LANE & 15);
  int mb = m0 + ((LANE >> 4) << 3);
  float bv = bias ? bias[n] : 0.0f;
#pragma unroll
  for (int i = 0; i < 8; ++i) C[(size_t)(mb + i) * ldc + n] = acc[i] + bv;
}

// ---------------- prep kernels ---------------------------------------------

__global__ void k_f32_to_bf16(const float* __restrict__ src, bf16_t* __restrict__ dst, int n) {
  int i = blockIdx.x * blockDim.x + threadIdx.x;
  if (i < n) dst[i] = (bf16_t)src[i];
}

__global__ void k_embed(const int* __restrict__ ids, const float* __restrict__ table,
                        bf16_t* __restrict__ out) {
  int r  = blockIdx.x;           // 0..T*B-1
  int id = ids[r];
  for (int e = threadIdx.x; e < 512; e += blockDim.x)
    out[(size_t)r * 512 + e] = (id == 0) ? (bf16_t)0.0f
                                         : (bf16_t)table[(size_t)id * 512 + e];
}

__global__ void k_ctx_bse(const float* __restrict__ ctx /*(S,B,He)*/,
                          bf16_t* __restrict__ out /*(B,S,He)*/) {
  int bs = blockIdx.x;           // b*S + s
  int b = bs >> 7, s = bs & 127;
  for (int e = threadIdx.x; e < 512; e += blockDim.x)
    out[((size_t)b * 128 + s) * 512 + e] = (bf16_t)ctx[((size_t)s * 64 + b) * 512 + e];
}

__global__ void k_init_state(const float* __restrict__ hidden, const float* __restrict__ init_att,
                             float* hf, bf16_t* hbf, float* cf, bf16_t* cbf) {
  int i = blockIdx.x * blockDim.x + threadIdx.x;
  if (i < 64 * 512) {
    float h = hidden[i], c = init_att[i];
    hf[i] = h; hbf[i] = (bf16_t)h;
    cf[i] = c; cbf[i] = (bf16_t)c;
  }
}

// ---------------- precompute = ctx_bse @ W_pre^T + b_pre (8192x512, K=512) --

__global__ __launch_bounds__(256)
void k_precompute(const bf16_t* __restrict__ A, const bf16_t* __restrict__ W,
                  const float* __restrict__ bias, float* __restrict__ C) {
  int m0 = blockIdx.x * 64 + (WAVE & 3) * 16;     // gridDim.x = 128
  int n0 = blockIdx.y * 128 + (WAVE >> 2) * 64;   // gridDim.y = 4
  v8f acc[4] = {};
  for (int k0 = 0; k0 < 512; k0 += 32) {
    v16bf a = load_a(A, 512, m0, k0);
#pragma unroll
    for (int j = 0; j < 4; ++j)
      acc[j] = wmma_bf16(a, load_b(W, 512, n0 + j * 16, k0), acc[j]);
  }
#pragma unroll
  for (int j = 0; j < 4; ++j) store_tile(C, 512, m0, n0 + j * 16, acc[j], bias);
}

// ---------------- per-step GRU: h1 = GRU([emb,ctx], h) ---------------------
// grid = 8 blocks x 256 threads; block b owns output cols [64b, 64b+64).
// All three gates accumulated in a single K-sweep per input segment so each
// A-fragment load feeds 6 WMMAs (r/z/n x 2 tiles).

__global__ __launch_bounds__(256)
void k_gru(const bf16_t* __restrict__ emb_t, const bf16_t* __restrict__ ctx_prev,
           const bf16_t* __restrict__ h_prev, const float* __restrict__ h_prev_f,
           const bf16_t* __restrict__ Wih /*1536x1024*/, const bf16_t* __restrict__ Whh /*1536x512*/,
           const float* __restrict__ bih, const float* __restrict__ bhh,
           float* __restrict__ h1_f, bf16_t* __restrict__ h1_bf) {
  int jc  = blockIdx.x * 64;
  int m0  = (WAVE & 3) * 16;
  int nb0 = jc + (WAVE >> 2) * 32;   // global col base of this wave's 2 tiles
  int rr = nb0, rz = 512 + nb0, rn = 1024 + nb0;  // W row bases per gate

  v8f ar[2] = {}, az[2] = {}, gi[2] = {}, gh[2] = {};

  // x = [emb, ctx]: emb against W_ih[:, 0:512]
  for (int k0 = 0; k0 < 512; k0 += 32) {
    v16bf a = load_a(emb_t, 512, m0, k0);
    ar[0] = wmma_bf16(a, load_b(Wih, 1024, rr,      k0), ar[0]);
    ar[1] = wmma_bf16(a, load_b(Wih, 1024, rr + 16, k0), ar[1]);
    az[0] = wmma_bf16(a, load_b(Wih, 1024, rz,      k0), az[0]);
    az[1] = wmma_bf16(a, load_b(Wih, 1024, rz + 16, k0), az[1]);
    gi[0] = wmma_bf16(a, load_b(Wih, 1024, rn,      k0), gi[0]);
    gi[1] = wmma_bf16(a, load_b(Wih, 1024, rn + 16, k0), gi[1]);
  }
  // ctx against W_ih[:, 512:1024]
  for (int k0 = 0; k0 < 512; k0 += 32) {
    v16bf a = load_a(ctx_prev, 512, m0, k0);
    ar[0] = wmma_bf16(a, load_b(Wih, 1024, rr,      512 + k0), ar[0]);
    ar[1] = wmma_bf16(a, load_b(Wih, 1024, rr + 16, 512 + k0), ar[1]);
    az[0] = wmma_bf16(a, load_b(Wih, 1024, rz,      512 + k0), az[0]);
    az[1] = wmma_bf16(a, load_b(Wih, 1024, rz + 16, 512 + k0), az[1]);
    gi[0] = wmma_bf16(a, load_b(Wih, 1024, rn,      512 + k0), gi[0]);
    gi[1] = wmma_bf16(a, load_b(Wih, 1024, rn + 16, 512 + k0), gi[1]);
  }
  // h against W_hh; r/z fold gh into same accumulators, n keeps gh separate
  for (int k0 = 0; k0 < 512; k0 += 32) {
    v16bf a = load_a(h_prev, 512, m0, k0);
    ar[0] = wmma_bf16(a, load_b(Whh, 512, rr,      k0), ar[0]);
    ar[1] = wmma_bf16(a, load_b(Whh, 512, rr + 16, k0), ar[1]);
    az[0] = wmma_bf16(a, load_b(Whh, 512, rz,      k0), az[0]);
    az[1] = wmma_bf16(a, load_b(Whh, 512, rz + 16, k0), az[1]);
    gh[0] = wmma_bf16(a, load_b(Whh, 512, rn,      k0), gh[0]);
    gh[1] = wmma_bf16(a, load_b(Whh, 512, rn + 16, k0), gh[1]);
  }

  int mb = m0 + ((LANE >> 4) << 3);
#pragma unroll
  for (int tt = 0; tt < 2; ++tt) {
    int hcol = nb0 + tt * 16 + (LANE & 15);       // 0..511 (global h column)
    float br_ = bih[hcol]        + bhh[hcol];
    float bz_ = bih[512 + hcol]  + bhh[512 + hcol];
    float bi_ = bih[1024 + hcol];
    float bh_ = bhh[1024 + hcol];
#pragma unroll
    for (int i = 0; i < 8; ++i) {
      float rv = 1.0f / (1.0f + __expf(-(ar[tt][i] + br_)));
      float zv = 1.0f / (1.0f + __expf(-(az[tt][i] + bz_)));
      float nv = fast_tanh(gi[tt][i] + bi_ + rv * (gh[tt][i] + bh_));
      float hp = h_prev_f[(size_t)(mb + i) * 512 + hcol];
      float h1 = (1.0f - zv) * nv + zv * hp;
      h1_f[(size_t)(mb + i) * 512 + hcol] = h1;
      h1_bf[(size_t)(mb + i) * 512 + hcol] = (bf16_t)h1;
    }
  }
}

// ---------------- q = h1 @ W_q^T (64x512, K=512); grid = 4 x 256 -----------

__global__ __launch_bounds__(256)
void k_q(const bf16_t* __restrict__ h1, const bf16_t* __restrict__ Wq,
         float* __restrict__ q) {
  int m0 = (WAVE & 3) * 16;
  int n0 = blockIdx.x * 128 + (WAVE >> 2) * 64;
  v8f acc[4] = {};
  for (int k0 = 0; k0 < 512; k0 += 32) {
    v16bf a = load_a(h1, 512, m0, k0);
#pragma unroll
    for (int j = 0; j < 4; ++j)
      acc[j] = wmma_bf16(a, load_b(Wq, 512, n0 + j * 16, k0), acc[j]);
  }
#pragma unroll
  for (int j = 0; j < 4; ++j) store_tile(q, 512, m0, n0 + j * 16, acc[j], nullptr);
}

// ---------------- attention: energy->softmax->new_ctx; grid = 64 (1/b) -----

__global__ __launch_bounds__(256)
void k_attn(const float* __restrict__ q, const float* __restrict__ pre,
            const float* __restrict__ wv, const float* __restrict__ mask,
            const float* __restrict__ ctx_sbe /* (S,B,He) f32 */,
            float* __restrict__ ctx_f, bf16_t* __restrict__ ctx_b,
            float* __restrict__ score_out) {
  __shared__ float qs[512];
  __shared__ float wvs[512];
  __shared__ float en[128];
  int b = blockIdx.x;
  for (int i = threadIdx.x; i < 512; i += 256) { qs[i] = q[b * 512 + i]; wvs[i] = wv[i]; }
  __syncthreads();

  int w = WAVE, lane = LANE;
  for (int s = w; s < 128; s += 8) {
    const float* p = pre + ((size_t)b * 128 + s) * 512;
    float part = 0.0f;
#pragma unroll
    for (int a0 = 0; a0 < 16; ++a0) {
      int a = lane * 16 + a0;
      part += fast_tanh(p[a] + qs[a]) * wvs[a];
    }
#pragma unroll
    for (int off = 16; off; off >>= 1) part += __shfl_xor(part, off, 32);
    if (lane == 0) {
      float m = mask[b * 128 + s];
      en[s] = part * (1.0f - m) + m * (-1e6f);
    }
  }
  __syncthreads();

  if (w == 0) {                       // softmax over 128 by wave 0
    float mx = -3.4e38f;
#pragma unroll
    for (int i = 0; i < 4; ++i) mx = fmaxf(mx, en[lane * 4 + i]);
#pragma unroll
    for (int off = 16; off; off >>= 1) mx = fmaxf(mx, __shfl_xor(mx, off, 32));
    float ex[4]; float sum = 0.0f;
#pragma unroll
    for (int i = 0; i < 4; ++i) { ex[i] = __expf(en[lane * 4 + i] - mx); sum += ex[i]; }
#pragma unroll
    for (int off = 16; off; off >>= 1) sum += __shfl_xor(sum, off, 32);
    float inv = 1.0f / sum;
#pragma unroll
    for (int i = 0; i < 4; ++i) en[lane * 4 + i] = ex[i] * inv;
  }
  __syncthreads();

  for (int s = threadIdx.x; s < 128; s += 256) score_out[b * 128 + s] = en[s];

  for (int e = threadIdx.x; e < 512; e += 256) {
    float c = 0.0f;
    for (int s = 0; s < 128; ++s) c += en[s] * ctx_sbe[((size_t)s * 64 + b) * 512 + e];
    ctx_f[b * 512 + e] = c;
    ctx_b[b * 512 + e] = (bf16_t)c;
  }
}

// ---------------- readout + maxout; grid = 4 x 256 -------------------------

__global__ __launch_bounds__(256)
void k_readout(const bf16_t* __restrict__ emb_t, const bf16_t* __restrict__ h1,
               const bf16_t* __restrict__ ctx, const bf16_t* __restrict__ Wr /*512x1536*/,
               const float* __restrict__ br, float* __restrict__ out_t /*64x256*/) {
  int lane = LANE;
  int m0 = (WAVE & 3) * 16;
  int n0 = blockIdx.x * 128 + (WAVE >> 2) * 64;
  v8f acc[4] = {};
  for (int seg = 0; seg < 3; ++seg) {
    const bf16_t* A = (seg == 0) ? emb_t : (seg == 1 ? h1 : ctx);
    for (int k0 = 0; k0 < 512; k0 += 32) {
      v16bf a = load_a(A, 512, m0, k0);
#pragma unroll
      for (int j = 0; j < 4; ++j)
        acc[j] = wmma_bf16(a, load_b(Wr, 1536, n0 + j * 16, seg * 512 + k0), acc[j]);
    }
  }
  int mb = m0 + ((lane >> 4) << 3);
#pragma unroll
  for (int j = 0; j < 4; ++j) {
    int n = n0 + j * 16 + (lane & 15);
    float bias = br[n];
#pragma unroll
    for (int i = 0; i < 8; ++i) {
      float v = acc[j][i] + bias;
      float o = __shfl_xor(v, 1, 32);   // maxout partner (n^1) lives in lane^1
      if ((n & 1) == 0) out_t[(size_t)(mb + i) * 256 + (n >> 1)] = fmaxf(v, o);
    }
  }
}

// ---------------- finalize: hid, scores[-1], cur_ctx -----------------------

__global__ void k_final(const float* __restrict__ hf, const float* __restrict__ cf,
                        const float* __restrict__ score,
                        float* __restrict__ out_hid, float* __restrict__ out_score,
                        float* __restrict__ out_ctx) {
  int i = blockIdx.x * blockDim.x + threadIdx.x;
  if (i < 64 * 512) { out_hid[i] = hf[i]; out_ctx[i] = cf[i]; }
  if (i < 64 * 128) out_score[i] = score[i];
}

// ---------------- host ------------------------------------------------------

extern "C" void kernel_launch(void* const* d_in, const int* in_sizes, int n_in,
                              void* d_out, int out_size, void* d_ws, size_t ws_size,
                              hipStream_t stream) {
  const int*   input    = (const int*)d_in[0];
  const float* hidden   = (const float*)d_in[1];
  const float* context  = (const float*)d_in[2];
  const float* mask     = (const float*)d_in[3];
  const float* init_att = (const float*)d_in[4];
  const float* table    = (const float*)d_in[5];
  const float* W_ih     = (const float*)d_in[6];
  const float* W_hh     = (const float*)d_in[7];
  const float* b_ih     = (const float*)d_in[8];
  const float* b_hh     = (const float*)d_in[9];
  const float* W_pre    = (const float*)d_in[10];
  const float* b_pre    = (const float*)d_in[11];
  const float* W_q      = (const float*)d_in[12];
  const float* w_v      = (const float*)d_in[13];
  const float* W_read   = (const float*)d_in[14];
  const float* b_read   = (const float*)d_in[15];

  char* ws = (char*)d_ws;
  size_t off = 0;
  auto alloc = [&](size_t bytes) -> char* {
    char* p = ws + off;
    off = (off + bytes + 255) & ~(size_t)255;
    return p;
  };

  bf16_t* wih  = (bf16_t*)alloc((size_t)1536 * 1024 * 2);
  bf16_t* whh  = (bf16_t*)alloc((size_t)1536 * 512 * 2);
  bf16_t* wq   = (bf16_t*)alloc((size_t)512 * 512 * 2);
  bf16_t* wrd  = (bf16_t*)alloc((size_t)512 * 1536 * 2);
  bf16_t* wpre = (bf16_t*)alloc((size_t)512 * 512 * 2);
  bf16_t* embb = (bf16_t*)alloc((size_t)64 * 64 * 512 * 2);
  bf16_t* ctxb = (bf16_t*)alloc((size_t)64 * 128 * 512 * 2);
  float*  pre  = (float*)alloc((size_t)64 * 128 * 512 * 4);
  bf16_t* hbf[2]  = { (bf16_t*)alloc(64 * 512 * 2), (bf16_t*)alloc(64 * 512 * 2) };
  bf16_t* cbf[2]  = { (bf16_t*)alloc(64 * 512 * 2), (bf16_t*)alloc(64 * 512 * 2) };
  float*  hf[2]   = { (float*)alloc(64 * 512 * 4),  (float*)alloc(64 * 512 * 4)  };
  float*  cf[2]   = { (float*)alloc(64 * 512 * 4),  (float*)alloc(64 * 512 * 4)  };
  float*  qbuf    = (float*)alloc(64 * 512 * 4);
  float*  scoreb  = (float*)alloc(64 * 128 * 4);

  float* out       = (float*)d_out;
  float* out_hid   = out + 64 * 64 * 256;            // 1048576
  float* out_score = out_hid + 64 * 512;             // 1081344
  float* out_ctx   = out_score + 64 * 128;           // 1089536

  // --- prep: bf16 weight copies, embedding gather, context transpose -------
  k_f32_to_bf16<<<(1536 * 1024 + 255) / 256, 256, 0, stream>>>(W_ih,   wih,  1536 * 1024);
  k_f32_to_bf16<<<(1536 * 512  + 255) / 256, 256, 0, stream>>>(W_hh,   whh,  1536 * 512);
  k_f32_to_bf16<<<(512 * 512   + 255) / 256, 256, 0, stream>>>(W_q,    wq,   512 * 512);
  k_f32_to_bf16<<<(512 * 1536  + 255) / 256, 256, 0, stream>>>(W_read, wrd,  512 * 1536);
  k_f32_to_bf16<<<(512 * 512   + 255) / 256, 256, 0, stream>>>(W_pre,  wpre, 512 * 512);
  k_embed<<<64 * 64, 256, 0, stream>>>(input, table, embb);
  k_ctx_bse<<<64 * 128, 256, 0, stream>>>(context, ctxb);
  k_init_state<<<(64 * 512 + 255) / 256, 256, 0, stream>>>(hidden, init_att,
                                                           hf[0], hbf[0], cf[0], cbf[0]);
  // precompute (B*S=8192, A=512, K=512)
  k_precompute<<<dim3(128, 4), 256, 0, stream>>>(ctxb, wpre, b_pre, pre);

  // --- sequential scan over T=64 steps -------------------------------------
  for (int t = 0; t < 64; ++t) {
    int cur = t & 1, nxt = cur ^ 1;
    const bf16_t* emb_t = embb + (size_t)t * 64 * 512;
    k_gru<<<8, 256, 0, stream>>>(emb_t, cbf[cur], hbf[cur], hf[cur],
                                 wih, whh, b_ih, b_hh, hf[nxt], hbf[nxt]);
    k_q<<<4, 256, 0, stream>>>(hbf[nxt], wq, qbuf);
    k_attn<<<64, 256, 0, stream>>>(qbuf, pre, w_v, mask, context,
                                   cf[nxt], cbf[nxt], scoreb);
    k_readout<<<4, 256, 0, stream>>>(emb_t, hbf[nxt], cbf[nxt], wrd, b_read,
                                     out + (size_t)t * 64 * 256);
  }

  // after t=63, state parity lands in buffer 0
  k_final<<<(64 * 512 + 255) / 256, 256, 0, stream>>>(hf[0], cf[0], scoreb,
                                                      out_hid, out_score, out_ctx);
  (void)in_sizes; (void)n_in; (void)out_size; (void)ws_size;
}
